// PREFFFT_85839216377942
// MI455X (gfx1250) — compile-verified, matrix operand order checked by reading, not run on hardware
//
#include <hip/hip_runtime.h>
#include <hip/hip_bf16.h>
#include <math.h>

// ---------------------------------------------------------------------------
// PREFFFT on MI455X (gfx1250):
//   masked ifft2  ==>  complex GEMMs on v_wmma_f32_16x16x32_f16
//   grid_sample + Fourier contraction ==> L2-resident f16 gather kernel
// ---------------------------------------------------------------------------

typedef __attribute__((ext_vector_type(16))) _Float16 v16h;
typedef __attribute__((ext_vector_type(8)))  _Float16 v8h;
typedef __attribute__((ext_vector_type(8)))  float    v8f;

#define RESV   256
#define CHV    16
#define RRV    8
#define MASKV  128
#define NPTS   262144

// ---- WMMA helper ----------------------------------------------------------
__device__ __forceinline__ v8f wmma_f16(v16h a, v16h b, v8f c) {
    // (neg_a, A, neg_b, B, c_mod, C, reuse_a, reuse_b)
    return __builtin_amdgcn_wmma_f32_16x16x32_f16(false, a, false, b,
                                                  (short)0, c, false, false);
}

// Load a 16x32 f16 fragment (A-layout; B uses the same layout from an N x K
// row-major matrix).  ISA 7.12.2, 16-bit A-matrix 16x32:
//   lanes 0-15 : M = lane,   VGPR0..3 -> K=0..7,  VGPR4..7 -> K=16..23
//   lanes16-31 : M = lane-16,VGPR0..3 -> K=8..15, VGPR4..7 -> K=24..31
__device__ __forceinline__ v16h frag_load(const _Float16* base, int ld,
                                          int row0, int k0, int lane) {
    int g = lane >> 4;
    int m = lane & 15;
    const _Float16* rp = base + (size_t)(row0 + m) * ld + k0 + g * 8;
    const uint4* q = reinterpret_cast<const uint4*>(rp);   // 16B aligned
    uint4 lo = q[0];          // halves K = g*8+0 .. g*8+7
    uint4 hi = q[2];          // halves K = g*8+16 .. g*8+23
    union { v16h v; uint32_t u[8]; } U;
    U.u[0] = lo.x; U.u[1] = lo.y; U.u[2] = lo.z; U.u[3] = lo.w;
    U.u[4] = hi.x; U.u[5] = hi.y; U.u[6] = hi.z; U.u[7] = hi.w;
    return U.v;
}

// ---- K0: build DFT twiddle tables  A[y][u] = exp(+2*pi*i*y*u/256) ---------
__global__ void PREFFFT_tables(_Float16* Are, _Float16* Aim, _Float16* Aimn) {
    int idx = blockIdx.x * blockDim.x + threadIdx.x;
    if (idx >= 256 * 128) return;
    int y = idx >> 7;
    int u = idx & 127;
    int ph = (y * u) & 255;                       // exact period-256 reduction
    float ang = (float)ph * (6.283185307179586f / 256.0f);
    float s, c;
    __sincosf(ang, &s, &c);
    Are[idx]  = (_Float16)c;
    Aim[idx]  = (_Float16)s;
    Aimn[idx] = (_Float16)(-s);
}

// ---- K1: stage masked spectrum quadrant, transposed (N=v rows, K=u cols) --
__global__ void PREFFFT_convert(const float* __restrict__ P,
                                _Float16* __restrict__ Xre,
                                _Float16* __restrict__ Xim) {
    int idx = blockIdx.x * blockDim.x + threadIdx.x;
    if (idx >= 128 * 128 * 128) return;
    int v = idx & 127;
    int u = (idx >> 7) & 127;
    int c = idx >> 14;
    const float* src = P + (((size_t)c * 256 + u) * 256 + v) * 2;
    size_t dst = ((size_t)c * 128 + v) * 128 + u;
    Xre[dst] = (_Float16)src[0];
    Xim[dst] = (_Float16)src[1];
}

// ---- K2: T[c](256x128) = A(256x128) * X[c](128x128), complex --------------
__global__ void PREFFFT_gemm1(const _Float16* __restrict__ Are,
                              const _Float16* __restrict__ Aim,
                              const _Float16* __restrict__ Aimn,
                              const _Float16* __restrict__ Xre,
                              const _Float16* __restrict__ Xim,
                              _Float16* __restrict__ Tre,
                              _Float16* __restrict__ Tim) {
    int lane = threadIdx.x & 31;
    int gw = blockIdx.x * (blockDim.x >> 5) + (threadIdx.x >> 5);
    int c   = gw >> 7;          // 16 m-tiles * 8 n-tiles per channel
    int rem = gw & 127;
    int mt  = rem >> 3;
    int nt  = rem & 7;

    const _Float16* xr = Xre + (size_t)c * 16384;
    const _Float16* xi = Xim + (size_t)c * 16384;

    v8f cre = {};
    v8f cim = {};
#pragma unroll
    for (int kk = 0; kk < 4; ++kk) {
        int k0 = kk * 32;
        v16h ar = frag_load(Are,  128, mt * 16, k0, lane);
        v16h ai = frag_load(Aim,  128, mt * 16, k0, lane);
        v16h an = frag_load(Aimn, 128, mt * 16, k0, lane);
        v16h br = frag_load(xr,   128, nt * 16, k0, lane);
        v16h bi = frag_load(xi,   128, nt * 16, k0, lane);
        cre = wmma_f16(ar, br, cre);   // +Re*Re
        cre = wmma_f16(an, bi, cre);   // -Im*Im
        cim = wmma_f16(ar, bi, cim);   // +Re*Im
        cim = wmma_f16(ai, br, cim);   // +Im*Re
    }

    int g = lane >> 4;
    int n = lane & 15;
    size_t base = ((size_t)c * 256 + mt * 16 + g * 8) * 128 + nt * 16 + n;
#pragma unroll
    for (int r = 0; r < 8; ++r) {
        Tre[base + (size_t)r * 128] = (_Float16)cre[r];
        Tim[base + (size_t)r * 128] = (_Float16)cim[r];
    }
}

// ---- K3: F[c](256x256) = T[c](256x128) * B(128x256); write permuted feat --
__global__ void PREFFFT_gemm2(const _Float16* __restrict__ Tre,
                              const _Float16* __restrict__ Tim,
                              const _Float16* __restrict__ Are,
                              const _Float16* __restrict__ Aim,
                              const _Float16* __restrict__ Aimn,
                              _Float16* __restrict__ feat) {
    int lane = threadIdx.x & 31;
    int gw = blockIdx.x * (blockDim.x >> 5) + (threadIdx.x >> 5);
    int c   = gw >> 8;          // 16 m-tiles * 16 n-tiles per channel
    int rem = gw & 255;
    int mt  = rem >> 4;
    int nt  = rem & 15;

    const _Float16* tr = Tre + (size_t)c * 32768;
    const _Float16* ti = Tim + (size_t)c * 32768;

    v8f fre = {};
    v8f fim = {};
#pragma unroll
    for (int kk = 0; kk < 4; ++kk) {
        int k0 = kk * 32;
        v16h lr = frag_load(tr,   128, mt * 16, k0, lane);
        v16h li = frag_load(ti,   128, mt * 16, k0, lane);
        // B^T[x][u] = A[x][u]: same table, rows indexed by output x
        v16h br = frag_load(Are,  128, nt * 16, k0, lane);
        v16h bi = frag_load(Aim,  128, nt * 16, k0, lane);
        v16h bn = frag_load(Aimn, 128, nt * 16, k0, lane);
        fre = wmma_f16(lr, br, fre);
        fre = wmma_f16(li, bn, fre);
        fim = wmma_f16(lr, bi, fim);
        fim = wmma_f16(li, br, fim);
    }

    // feat channel permutation: original f = 2c+im ->
    //   part = f>>7, fl = f&127, co = fl>>3, r = fl&7, f' = r*32 + part*16 + co
    int f0 = 2 * c,     p0 = f0 >> 7, l0 = f0 & 127;
    int f1 = 2 * c + 1, p1 = f1 >> 7, l1 = f1 & 127;
    int fp0 = (l0 & 7) * 32 + p0 * 16 + (l0 >> 3);
    int fp1 = (l1 & 7) * 32 + p1 * 16 + (l1 >> 3);

    const float sc = 1.0f / 65536.0f;   // ifft2 normalization
    int g = lane >> 4;
    int x = nt * 16 + (lane & 15);
#pragma unroll
    for (int r = 0; r < 8; ++r) {
        int y = mt * 16 + r + g * 8;
        size_t px = ((size_t)y * 256 + x) * 256;
        feat[px + fp0] = (_Float16)(fre[r] * sc);
        feat[px + fp1] = (_Float16)(fim[r] * sc);
    }
}

// ---- K4: bilinear gather + Fourier contraction (L2-resident) --------------
__global__ void PREFFFT_sample(const float* __restrict__ inp,
                               const _Float16* __restrict__ feat,
                               float* __restrict__ out) {
    int n = blockIdx.x * blockDim.x + threadIdx.x;
    if (n >= NPTS) return;

    float i0 = inp[(size_t)n * 3 + 0];
    float i1 = inp[(size_t)n * 3 + 1];
    float i2 = inp[(size_t)n * 3 + 2];
    float gxs[3] = { i1, i0, i0 };
    float gys[3] = { i2, i2, i1 };
    float ics[3] = { i0, i1, i2 };

    float acc[16];
#pragma unroll
    for (int j = 0; j < 16; ++j) acc[j] = 0.0f;

#pragma unroll
    for (int p = 0; p < 3; ++p) {
        float x = (gxs[p] + 1.0f) * 0.5f * 255.0f;
        float y = (gys[p] + 1.0f) * 0.5f * 255.0f;
        float fx = floorf(x), fy = floorf(y);
        float wx = x - fx,    wy = y - fy;
        int x0 = min(max((int)fx, 0), 255);
        int x1 = min(x0 + 1, 255);
        int y0 = min(max((int)fy, 0), 255);
        int y1 = min(y0 + 1, 255);
        float w00 = (1.0f - wx) * (1.0f - wy);
        float w01 = wx * (1.0f - wy);
        float w10 = (1.0f - wx) * wy;
        float w11 = wx * wy;

        float s = (ics[p] + 1.0f) * 0.5f * 255.0f;
        float cs[8], sn[8];
#pragma unroll
        for (int r = 0; r < 8; ++r) {
            float ang = s * (6.283185307179586f / 256.0f) * (float)((1 << r) - 1);
            __sincosf(ang, &sn[r], &cs[r]);
        }

        const _Float16* fb = feat + (size_t)p * 16777216;
        const _Float16* tp[4] = {
            fb + ((size_t)y0 * 256 + x0) * 256,
            fb + ((size_t)y0 * 256 + x1) * 256,
            fb + ((size_t)y1 * 256 + x0) * 256,
            fb + ((size_t)y1 * 256 + x1) * 256
        };
        float tw[4] = { w00, w01, w10, w11 };

#pragma unroll
        for (int t = 0; t < 4; ++t) {
            __builtin_prefetch(tp[t], 0, 1);
            const v8h* q = reinterpret_cast<const v8h*>(tp[t]);
            float wt = tw[t];
#pragma unroll
            for (int r = 0; r < 8; ++r) {
                v8h a0 = q[r * 4 + 0];   // real, co 0..7
                v8h a1 = q[r * 4 + 1];   // real, co 8..15
                v8h a2 = q[r * 4 + 2];   // imag, co 0..7
                v8h a3 = q[r * 4 + 3];   // imag, co 8..15
                float wc = wt * cs[r];
                float ws = wt * sn[r];
#pragma unroll
                for (int j = 0; j < 8; ++j) {
                    acc[j]     += wc * (float)a0[j] - ws * (float)a2[j];
                    acc[8 + j] += wc * (float)a1[j] - ws * (float)a3[j];
                }
            }
        }
    }

#pragma unroll
    for (int co = 0; co < 16; ++co)
        out[(size_t)n * 16 + co] = acc[co];
}

// ---------------------------------------------------------------------------
extern "C" void kernel_launch(void* const* d_in, const int* in_sizes, int n_in,
                              void* d_out, int out_size, void* d_ws, size_t ws_size,
                              hipStream_t stream) {
    const float* P[3] = { (const float*)d_in[0],
                          (const float*)d_in[1],
                          (const float*)d_in[2] };
    const float* inp = (const float*)d_in[3];
    float* out = (float*)d_out;

    char* ws = (char*)d_ws;
    const size_t featHalves = (size_t)3 * 256 * 256 * 256;      // 96 MB f16
    _Float16* feat = (_Float16*)ws;
    _Float16* Are  = (_Float16*)(ws + featHalves * sizeof(_Float16));
    _Float16* Aim  = Are  + 256 * 128;
    _Float16* Aimn = Aim  + 256 * 128;
    _Float16* Xre  = Aimn + 256 * 128;
    _Float16* Xim  = Xre  + 128 * 128 * 128;
    _Float16* Tre  = Xim  + 128 * 128 * 128;
    _Float16* Tim  = Tre  + 128 * 256 * 128;

    PREFFFT_tables<<<128, 256, 0, stream>>>(Are, Aim, Aimn);

    for (int p = 0; p < 3; ++p) {
        PREFFFT_convert<<<8192, 256, 0, stream>>>(P[p], Xre, Xim);
        PREFFFT_gemm1<<<2048, 256, 0, stream>>>(Are, Aim, Aimn, Xre, Xim, Tre, Tim);
        PREFFFT_gemm2<<<4096, 256, 0, stream>>>(Tre, Tim, Are, Aim, Aimn,
                                                feat + (size_t)p * 16777216);
    }

    PREFFFT_sample<<<1024, 256, 0, stream>>>(inp, feat, out);
}
